// DeepSet_51384988729811
// MI455X (gfx1250) — matrix operand order repaired and checked by reading, not compiled, not run on hardware
//
#include <hip/hip_runtime.h>
#include <hip/hip_bf16.h>

// ---------------------------------------------------------------------------
// Fused DeepSets energy network for MI455X (gfx1250, wave32, WMMA).
// Each wave processes 16 points (one WMMA M-tile) through all 33 linear
// layers using v_wmma_f32_16x16x32_f16, ping-ponging f16 activations through
// wave-private LDS buffers. Weights are staged once per workgroup into LDS in
// B-operand layout ([n][kpad] f16, K zero-padded to multiples of 32).
// All per-WMMA guards are branchless (address select into an LDS zero pad).
// ---------------------------------------------------------------------------

typedef __attribute__((ext_vector_type(16))) _Float16 v16h;
typedef __attribute__((ext_vector_type(8)))  _Float16 v8h;
typedef __attribute__((ext_vector_type(4)))  _Float16 v4h;
typedef __attribute__((ext_vector_type(8)))  float    v8f;

#define NW   4      // waves per workgroup
#define TPW  8      // 16-point tiles per wave
#define BW   64     // activation row width in halves (covers K up to 64)
#define BLOCK_THREADS (NW * 32)

// ---- per-layer constant tables (33 WMMA layers, staged order) --------------
// order: b0 gam0..4, b0 lam1..4, b1 gam0..4, b1 lam1..4, b2 gam0..4,
//        b2 lam1..4, proj0..5
constexpr int LDIN[33]  = { 4,10,20,15,10,  10,20,15,10,
                           10,10,20,15,10,  10,20,15,10,
                           20,10,20,15,10,  10,20,15,10,
                           60,20,40,30,10,5 };
constexpr int LDOUT[33] = {10,20,15,10,10,  20,15,10,10,
                           10,20,15,10,20,  20,15,10,20,
                           10,20,15,10,30,  20,15,10,30,
                           20,40,30,10,5,1 };
constexpr int LKPAD[33] = {32,32,32,32,32,  32,32,32,32,
                           32,32,32,32,32,  32,32,32,32,
                           32,32,32,32,32,  32,32,32,32,
                           64,32,64,32,32,32 };
constexpr int LWOFF[33] = {0,320,960,1440,1760,
                           2080,2720,3200,3520,
                           3840,4160,4800,5280,5600,
                           6240,6880,7360,7680,
                           8320,8640,9280,9760,10080,
                           11040,11680,12160,12480,
                           13440,14720,16000,17920,18240,18400 };
// bias offsets, padded to ceil16(dout) with zeros (removes per-lane guard)
constexpr int LBOFF[33] = {0,16,48,64,80,
                           96,128,144,160,
                           176,192,224,240,256,
                           288,320,336,352,
                           384,400,432,448,464,
                           496,528,544,560,
                           592,624,672,704,720,736};
#define WZOFF     18432            // 64-half zero pad for out-of-range B rows
#define WLDS_SIZE (18432 + 64)
#define BLDS_SIZE 752

struct KArgs {
    const float* x;        // (P, 4)
    const int*   xidx;     // (P, 3)
    float*       out;      // (2048, 6, 64, 64)
    const float* W[33];    // layer weights (din, dout) row-major f32
    const float* Bb[33];   // layer biases (dout,)
    const float* l0W[3];   // lam layer-0 weights (1,10)
    const float* l0B[3];   // lam layer-0 biases (10,)
    int          npts;
};

// lrelu(x,s) == max(x, s*x) for any slope s < 1 (holds for -0.8 and 0.2)
__device__ __forceinline__ float lrelu(float v, float s) {
    return fmaxf(v, v * s);
}

// fast tanh on the hardware transcendental: 1 - 2/(e^{2x}+1)
// (saturates to +/-1 at the infinities, tanh_fast(0) == 0, no NaN)
__device__ __forceinline__ float tanh_fast(float x) {
    const float t = __expf(2.f * x);
    return 1.f - 2.f * __builtin_amdgcn_rcpf(t + 1.f);
}

// zero a 16xBW f16 buffer (2KB) with one wave: 4x b128 stores per lane
__device__ __forceinline__ void store_zero(_Float16* dst) {
    const int lane = threadIdx.x & 31;
    v8h z = {};
    v8h* d = reinterpret_cast<v8h*>(dst);
#pragma unroll
    for (int j = 0; j < 4; ++j) d[lane * 4 + j] = z;
}

// zero one 16-column group [C0, C0+16) across all 16 rows: 1 b128 per lane
template <int C0>
__device__ __forceinline__ void zero_cols16(_Float16* dst) {
    const int lane = threadIdx.x & 31;
    const int row  = lane >> 1;
    const int ch   = (lane & 1) * 8;
    v8h z = {};
    *reinterpret_cast<v8h*>(dst + row * BW + C0 + ch) = z;
}

// store C fragments (f32) as f16 activations; ZG selects whether one extra
// 16-column group must be zeroed so the next layer's K-read sees zeros.
template <int NT, int ZG>
__device__ __forceinline__ void store_act(_Float16* dst, const v8f c[3]) {
    if constexpr (ZG > 0) zero_cols16<NT * 16>(dst);
    const int lane = threadIdx.x & 31;
    const int nl   = lane & 15;
    const int mo   = (lane < 16) ? 0 : 8;
#pragma unroll
    for (int nt = 0; nt < NT; ++nt)
#pragma unroll
        for (int r = 0; r < 8; ++r)
            dst[(mo + r) * BW + nt * 16 + nl] = (_Float16)c[nt][r];
}

// store into the concat buffer at a column offset (no zero-fill; the concat
// buffer is zeroed once per tile and blocks are written in ascending order).
template <int NT, int CO>
__device__ __forceinline__ void store_cat(_Float16* dst, const v8f c[3]) {
    const int lane = threadIdx.x & 31;
    const int nl   = lane & 15;
    const int mo   = (lane < 16) ? 0 : 8;
#pragma unroll
    for (int nt = 0; nt < NT; ++nt)
#pragma unroll
        for (int r = 0; r < 8; ++r)
            dst[(mo + r) * BW + CO + nt * 16 + nl] = (_Float16)c[nt][r];
}

template <int NT>
__device__ __forceinline__ void act_lrelu(v8f c[3], float s) {
#pragma unroll
    for (int nt = 0; nt < NT; ++nt)
#pragma unroll
        for (int r = 0; r < 8; ++r) c[nt][r] = lrelu(c[nt][r], s);
}

template <int NT>
__device__ __forceinline__ void act_proj(v8f c[3]) {
#pragma unroll
    for (int nt = 0; nt < NT; ++nt)
#pragma unroll
        for (int r = 0; r < 8; ++r) c[nt][r] = tanh_fast(lrelu(c[nt][r], -0.8f));
}

// ---- one linear layer as WMMA: C(16 x dout) = A(16 x din) @ B(din x dout) --
// A source: LDS activations, rows of BW halves, K zero-padded.
// B source: staged LDS weights at LWOFF[LI], [n][kpad] layout, zero-padded;
// out-of-range columns read from the WZOFF zero pad (branchless addr select).
template <int LI>
__device__ __forceinline__ void layer(const _Float16* __restrict__ src,
                                      const _Float16* __restrict__ wl,
                                      const float* __restrict__ bl,
                                      v8f c[3]) {
    constexpr int din  = LDIN[LI];
    constexpr int dout = LDOUT[LI];
    constexpr int kp   = LKPAD[LI];
    constexpr int wo   = LWOFF[LI];
    constexpr int bo   = LBOFF[LI];
    constexpr int NT   = (dout + 15) / 16;
    constexpr int KT   = (din + 31) / 32;

    const int lane = threadIdx.x & 31;
    const int m    = lane & 15;
    const int ao   = (lane < 16) ? 0 : 8;    // A: K halves 0..7 / 8..15 (+16)
    const int bo2  = (lane < 16) ? 0 : 8;    // B: K rows 0..7 / 8..15 (+16)

#pragma unroll
    for (int nt = 0; nt < NT; ++nt) {
        const int col  = nt * 16 + m;
        const int wbase = (col < dout) ? (wo + col * kp) : WZOFF;
        v8f acc = {};
#pragma unroll
        for (int kt = 0; kt < KT; ++kt) {
            const v8h* ap =
                reinterpret_cast<const v8h*>(src + m * BW + kt * 32 + ao);
            v16h A = __builtin_shufflevector(ap[0], ap[2],
                0,1,2,3,4,5,6,7,8,9,10,11,12,13,14,15);
            const v8h* bp = reinterpret_cast<const v8h*>(
                wl + wbase + kt * 32 + bo2);
            v16h B = __builtin_shufflevector(bp[0], bp[2],
                0,1,2,3,4,5,6,7,8,9,10,11,12,13,14,15);
            acc = __builtin_amdgcn_wmma_f32_16x16x32_f16(
                false, A, false, B, (short)0, acc, false, false);
        }
        const float bias = bl[bo + col];     // zero-padded, no guard
#pragma unroll
        for (int r = 0; r < 8; ++r) acc[r] += bias;
        c[nt] = acc;
    }
}

// ---- one PermEqui block ----------------------------------------------------
template <int LG, int LL, int DINB, int DOUTB, int CATOFF, bool LAST>
__device__ __forceinline__ void phi_block(
    _Float16* b0, _Float16* b1, _Float16* cat, float* scr,
    const _Float16* wl, const float* bl, float l0w, float l0b) {
    const int lane = threadIdx.x & 31;
    const int mo   = (lane < 16) ? 0 : 8;

    // per-point feature mean (block input lives in b0)
    if (lane < 16) {
        float s = 0.f;
#pragma unroll
        for (int k = 0; k < DINB; ++k) s += (float)b0[lane * BW + k];
        scr[lane] = s * (1.f / (float)DINB);
    }
    float xmv[8];
#pragma unroll
    for (int r = 0; r < 8; ++r) xmv[r] = scr[mo + r];

    v8f t[3], cg[3], cl[3];
    // gam chain: DINB->10->20->15->10->DOUTB, lrelu(-0.8) between layers
    layer<LG + 0>(b0, wl, bl, t); act_lrelu<1>(t, -0.8f); store_act<1, 1>(b1, t);
    layer<LG + 1>(b1, wl, bl, t); act_lrelu<2>(t, -0.8f); store_act<2, 0>(b0, t);
    layer<LG + 2>(b0, wl, bl, t); act_lrelu<1>(t, -0.8f); store_act<1, 1>(b1, t);
    layer<LG + 3>(b1, wl, bl, t); act_lrelu<1>(t, -0.8f); store_act<1, 1>(b0, t);
    layer<LG + 4>(b0, wl, bl, cg);                 // kept in registers

    // lam chain: 1->10 done elementwise from xm, then 10->20->15->10->DOUTB
#pragma unroll
    for (int r = 0; r < 8; ++r)
        t[0][r] = lrelu(xmv[r] * l0w + l0b, -0.8f);
    store_act<1, 1>(b1, t);
    layer<LL + 0>(b1, wl, bl, t); act_lrelu<2>(t, -0.8f); store_act<2, 0>(b0, t);
    layer<LL + 1>(b0, wl, bl, t); act_lrelu<1>(t, -0.8f); store_act<1, 1>(b1, t);
    layer<LL + 2>(b1, wl, bl, t); act_lrelu<1>(t, -0.8f); store_act<1, 1>(b0, t);
    layer<LL + 3>(b0, wl, bl, cl);

    // h = lrelu(tanh(gam - lam), 0.2)
    constexpr int NT = (DOUTB + 15) / 16;
#pragma unroll
    for (int nt = 0; nt < NT; ++nt)
#pragma unroll
        for (int r = 0; r < 8; ++r) {
            float v = tanh_fast(cg[nt][r] - cl[nt][r]);
            cg[nt][r] = lrelu(v, 0.2f);
        }
    store_cat<NT, CATOFF>(cat, cg);
    if (!LAST) store_act<NT, (NT == 1 ? 1 : 0)>(b0, cg);  // next block input
}

__global__ __launch_bounds__(BLOCK_THREADS)
void deepset_fused(KArgs a) {
    __shared__ _Float16 WL[WLDS_SIZE];
    __shared__ float    BL[BLDS_SIZE];
    __shared__ _Float16 ACT[NW][3][16 * BW];   // buf0, buf1, concat per wave
    __shared__ float    SCR[NW][32];           // [0..15]=xm, [16..31]=energy

    const int tid = threadIdx.x;

    // ---- stage all weights/biases into LDS (f16, [n][kpad], zero-padded) ---
    for (int l = 0; l < 33; ++l) {
        const int din = LDIN[l], dout = LDOUT[l], kp = LKPAD[l];
        const int wo = LWOFF[l], bo = LBOFF[l];
        const float* W = a.W[l];
        for (int t = tid; t < dout * kp; t += BLOCK_THREADS) {
            const int n = t / kp;
            const int k = t - n * kp;
            WL[wo + t] = (k < din) ? (_Float16)W[k * dout + n] : (_Float16)0.f;
        }
        const float* Bv = a.Bb[l];
        const int pd = (dout + 15) & ~15;
        for (int t = tid; t < pd; t += BLOCK_THREADS)
            BL[bo + t] = (t < dout) ? Bv[t] : 0.f;
    }
    for (int t = tid; t < 64; t += BLOCK_THREADS)       // B zero-pad rows
        WL[WZOFF + t] = (_Float16)0.f;
    __syncthreads();

    const int lane = tid & 31;
    const int wave = tid >> 5;
    const int nl   = lane & 15;
    const int mo   = (lane < 16) ? 0 : 8;

    // per-lane lam layer-0 params, lane holds column nl
    float l0w[3], l0b[3];
#pragma unroll
    for (int b = 0; b < 3; ++b) {
        l0w[b] = (nl < 10) ? a.l0W[b][nl] : 0.f;
        l0b[b] = (nl < 10) ? a.l0B[b][nl] : 0.f;
    }

    _Float16* b0  = &ACT[wave][0][0];
    _Float16* b1  = &ACT[wave][1][0];
    _Float16* cat = &ACT[wave][2][0];
    float*    scr = &SCR[wave][0];

    const int base = blockIdx.x * (NW * TPW * 16) + wave * (TPW * 16);

    for (int tile = 0; tile < TPW; ++tile) {
        const int p0 = base + tile * 16;
        if (p0 + 16 > a.npts) break;          // wave-uniform

        // ---- load 16 points (4 features) into b0 as f16, zero-padded ------
        store_zero(cat);
        store_zero(b0);
        if (lane < 16) {
            const float4 xv =
                reinterpret_cast<const float4*>(a.x)[p0 + lane];
            v4h h4 = { (_Float16)xv.x, (_Float16)xv.y,
                       (_Float16)xv.z, (_Float16)xv.w };
            *reinterpret_cast<v4h*>(b0 + lane * BW) = h4;
        }

        // ---- three PermEqui blocks ----------------------------------------
        phi_block< 0,  5,  4, 10,  0, false>(b0, b1, cat, scr, WL, BL, l0w[0], l0b[0]);
        phi_block< 9, 14, 10, 20, 10, false>(b0, b1, cat, scr, WL, BL, l0w[1], l0b[1]);
        phi_block<18, 23, 20, 30, 30, true >(b0, b1, cat, scr, WL, BL, l0w[2], l0b[2]);

        // ---- projection MLP: 60->20->40->30->10->5->1 ---------------------
        v8f t[3];
        layer<27>(cat, WL, BL, t); act_proj<2>(t); store_act<2, 0>(b0, t);
        layer<28>(b0,  WL, BL, t); act_proj<3>(t); store_act<3, 1>(b1, t);
        layer<29>(b1,  WL, BL, t); act_proj<2>(t); store_act<2, 0>(b0, t);
        layer<30>(b0,  WL, BL, t); act_proj<1>(t); store_act<1, 1>(b1, t);
        layer<31>(b1,  WL, BL, t); act_proj<1>(t); store_act<1, 1>(b0, t);
        layer<32>(b0,  WL, BL, t);                 // (P,1) energy in t[0]

        // ---- scatter energies into the image ------------------------------
        if (nl == 0) {   // lanes 0 and 16 hold column n=0 for m=0..7 / 8..15
#pragma unroll
            for (int r = 0; r < 8; ++r) scr[16 + mo + r] = t[0][r];
        }
        if (lane < 16) {
            const float e = scr[16 + lane];
            const int p   = p0 + lane;
            const int* ip = a.xidx + 3 * p;
            const int c = ip[0], y = ip[1], z = ip[2];
            const int seq = p >> 9;                    // SEQ_LEN = 512
            if (c >= 1)
                a.out[((seq * 6 + (c - 1)) * 64 + y) * 64 + z] = e;
        }
    }
}

extern "C" void kernel_launch(void* const* d_in, const int* in_sizes, int n_in,
                              void* d_out, int out_size, void* d_ws, size_t ws_size,
                              hipStream_t stream) {
    (void)n_in; (void)d_ws; (void)ws_size;

    KArgs a;
    a.x    = (const float*)d_in[0];
    a.xidx = (const int*)d_in[73];
    a.out  = (float*)d_out;
    a.npts = in_sizes[0] / 4;

    // flatten {phi_params: [{gam: 5x(W,b), lam: 5x(W,b)} x3], proj: 6x(W,b)}
    int wi = 0;
    for (int b = 0; b < 3; ++b) {
        const int gbase = 1 + b * 20;
        for (int i = 0; i < 5; ++i) {           // gam layers
            a.W[wi]  = (const float*)d_in[gbase + 2 * i];
            a.Bb[wi] = (const float*)d_in[gbase + 2 * i + 1];
            ++wi;
        }
        a.l0W[b] = (const float*)d_in[gbase + 10];   // lam layer 0
        a.l0B[b] = (const float*)d_in[gbase + 11];
        for (int i = 1; i < 5; ++i) {           // lam layers 1..4
            a.W[wi]  = (const float*)d_in[gbase + 10 + 2 * i];
            a.Bb[wi] = (const float*)d_in[gbase + 10 + 2 * i + 1];
            ++wi;
        }
    }
    for (int j = 0; j < 6; ++j) {               // proj layers
        a.W[wi]  = (const float*)d_in[61 + 2 * j];
        a.Bb[wi] = (const float*)d_in[62 + 2 * j];
        ++wi;
    }

    // image must be zero except scattered energies
    hipMemsetAsync(d_out, 0, (size_t)out_size * sizeof(float), stream);

    const int ptsPerWG = NW * TPW * 16;         // 512
    const int grid = (a.npts + ptsPerWG - 1) / ptsPerWG;
    deepset_fused<<<grid, BLOCK_THREADS, 0, stream>>>(a);
}
